// NodeFeatures_5789615915409
// MI455X (gfx1250) — matrix operand order repaired and checked by reading, not compiled, not run on hardware
//
#include <hip/hip_runtime.h>
#include <hip/hip_bf16.h>

// B=8, N=2000, E=20, H=128
#define BB   8
#define NN   2000
#define EE   20
#define HH   128
#define NE   (NN * EE)          // 40000 edge rows per batch
#define BN   (BB * NN)          // 16000 node rows total
#define WSTR 136                // LDS row stride (bf16 elems) for W^T; 136*2=272B keeps 16B align, avoids bank conflicts

typedef __attribute__((ext_vector_type(16))) __bf16 v16bf;
typedef __attribute__((ext_vector_type(8)))  float  v8f;

static __device__ __forceinline__ v8f wmma_bf16(v16bf a, v16bf b, v8f c) {
    // D(16x16 f32) = A(16x32 bf16) x B(32x16 bf16) + C
    return __builtin_amdgcn_wmma_f32_16x16x32_bf16(false, a, false, b, (short)0, c, false, false);
}

// A-operand (16x32, bf16) lane layout per CDNA5 ISA 7.12.2:
//  lane&15 = row M; (lane>>4)*8 = K sub-base; elems 0..7 -> K=kb..kb+7, elems 8..15 -> K=kb+16..kb+23
static __device__ __forceinline__ v16bf load_a_bf16(const float* __restrict__ base, int ko, int lane) {
    const float* p = base + (size_t)(lane & 15) * HH + ko * 32 + ((lane >> 4) * 8);
    v16bf a;
#pragma unroll
    for (int i = 0; i < 8; ++i) {
        a[i]     = (__bf16)p[i];
        a[i + 8] = (__bf16)p[16 + i];
    }
    return a;
}

static __device__ __forceinline__ v16bf zero_a() {
    v16bf a;
#pragma unroll
    for (int i = 0; i < 16; ++i) a[i] = (__bf16)0.0f;
    return a;
}

// B-operand (32x16, bf16): lane&15 = column N; (lane>>4)*8 = K sub-base; elems 0..7 -> K=kb..kb+7, elems 8..15 -> K=kb+16..kb+23.
// wt holds W^T (row n = column of W), so each lane reads two runs of 8 consecutive bf16 (p[0..7], p[16..23]).
static __device__ __forceinline__ v16bf load_b_bf16(const __bf16* wt, int nt, int ko, int lane) {
    const __bf16* p = wt + (size_t)(nt * 16 + (lane & 15)) * WSTR + ko * 32 + ((lane >> 4) * 8);
    v16bf b;
#pragma unroll
    for (int i = 0; i < 8; ++i) { b[i] = p[i]; b[i + 8] = p[16 + i]; }
    return b;
}

// ---------------------------------------------------------------------------
// Kernel 1: Y[rows x 128] = X[rows x 128] @ W[128 x 128] + bias
// one wave -> one 16x128 strip; 8 waves per block; W^T(bf16) staged in LDS.
// ---------------------------------------------------------------------------
__global__ __launch_bounds__(256) void gemm128_bias_kernel(
    const float* __restrict__ X, const float* __restrict__ W,
    const float* __restrict__ bias, float* __restrict__ Y, int ntiles)
{
    __shared__ __bf16 wt[HH * WSTR];
    const int tid = threadIdx.x;
    for (int i = tid; i < HH * HH; i += 256) {
        int k = i >> 7, n = i & 127;
        wt[n * WSTR + k] = (__bf16)W[i];
    }
    __syncthreads();

    const int wave = tid >> 5, lane = tid & 31;
    const int tile = blockIdx.x * 8 + wave;
    if (tile >= ntiles) return;

    const float* base = X + (size_t)tile * 16 * HH;
    v8f acc[8];
#pragma unroll
    for (int nt = 0; nt < 8; ++nt)
#pragma unroll
        for (int g = 0; g < 8; ++g) acc[nt][g] = 0.0f;

#pragma unroll
    for (int ko = 0; ko < 4; ++ko) {
        v16bf a = load_a_bf16(base, ko, lane);
#pragma unroll
        for (int nt = 0; nt < 8; ++nt) {
            v16bf b = load_b_bf16(wt, nt, ko, lane);
            acc[nt] = wmma_bf16(a, b, acc[nt]);
        }
    }

    // C/D layout: VGPR g -> row M = g + 8*(lane>>4); column N = nt*16 + (lane&15)
    const int hi = lane >> 4, nlo = lane & 15;
#pragma unroll
    for (int nt = 0; nt < 8; ++nt) {
        float bv = bias[nt * 16 + nlo];
#pragma unroll
        for (int g = 0; g < 8; ++g) {
            int m = g + 8 * hi;
            Y[((size_t)tile * 16 + m) * HH + nt * 16 + nlo] = acc[nt][g] + bv;
        }
    }
}

// ---------------------------------------------------------------------------
// Kernel 2 (fused): per node n -- Ve = e@We (bf16 WMMA, 32x128 strip, rows>=20
// masked to zero), softmax over the 20 edges per channel (in registers, cross
// half-wave via shfl_xor 16), gather Vx via edge_index (L2-resident scratch),
// weighted sum, out += result (Ux already stored in d_out by kernel 1).
// be is constant along the edge axis and cancels in softmax -> skipped.
// One wave per node; 8 nodes per block.
// ---------------------------------------------------------------------------
__global__ __launch_bounds__(256) void edge_softmax_gather_kernel(
    const float* __restrict__ e, const float* __restrict__ We,
    const int* __restrict__ edge_index, const float* __restrict__ vx,
    float* __restrict__ out)
{
    __shared__ __bf16 wt[HH * WSTR];
    const int tid = threadIdx.x;
    for (int i = tid; i < HH * HH; i += 256) {
        int k = i >> 7, n = i & 127;
        wt[n * WSTR + k] = (__bf16)We[i];
    }
    __syncthreads();

    const int wave = tid >> 5, lane = tid & 31;
    const int b    = blockIdx.x / (NN / 8);
    const int node = (blockIdx.x % (NN / 8)) * 8 + wave;

    const float* ebase = e + ((size_t)b * NE + (size_t)node * EE) * HH;

    v8f acc[16];  // [mt*8 + nt]; mt=0 rows 0..15, mt=1 rows 16..31 (rows 20..31 are garbage/zero)
#pragma unroll
    for (int t = 0; t < 16; ++t)
#pragma unroll
        for (int g = 0; g < 8; ++g) acc[t][g] = 0.0f;

    const bool a1_valid = (lane & 15) < 4;  // rows 16..19 only
#pragma unroll
    for (int ko = 0; ko < 4; ++ko) {
        v16bf a0 = load_a_bf16(ebase, ko, lane);
        v16bf a1 = a1_valid ? load_a_bf16(ebase + 16 * HH, ko, lane) : zero_a();
#pragma unroll
        for (int nt = 0; nt < 8; ++nt) {
            v16bf bm = load_b_bf16(wt, nt, ko, lane);
            acc[nt]     = wmma_bf16(a0, bm, acc[nt]);
            acc[8 + nt] = wmma_bf16(a1, bm, acc[8 + nt]);
        }
    }

    // Softmax over edges j=0..19 per channel.
    // Lane (hi=0) holds rows {0..7} U {16..19}; lane (hi=1) holds rows {8..15}.
    const int hi = lane >> 4, nlo = lane & 15;
    const int* ixp = edge_index + b * NE + node * EE;
    int idx0[8];
#pragma unroll
    for (int g = 0; g < 8; ++g) idx0[g] = ixp[g + 8 * hi];
    int idx1[4];
#pragma unroll
    for (int g = 0; g < 4; ++g) idx1[g] = (hi == 0) ? ixp[16 + g] : 0;

    const float* vxb  = vx + (size_t)b * NN * HH;
    float*       orow = out + ((size_t)b * NN + node) * HH;

#pragma unroll
    for (int nt = 0; nt < 8; ++nt) {
        const int ch = nt * 16 + nlo;
        float mx = -3.4e38f;
#pragma unroll
        for (int g = 0; g < 8; ++g) mx = fmaxf(mx, acc[nt][g]);
        if (hi == 0) {
#pragma unroll
            for (int g = 0; g < 4; ++g) mx = fmaxf(mx, acc[8 + nt][g]);
        }
        mx = fmaxf(mx, __shfl_xor(mx, 16));   // full 20-row max, both half-lanes

        float s = 0.0f, ws = 0.0f;
#pragma unroll
        for (int g = 0; g < 8; ++g) {
            float p = __expf(acc[nt][g] - mx);
            s += p;
            ws += p * vxb[(size_t)idx0[g] * HH + ch];
        }
        if (hi == 0) {
#pragma unroll
            for (int g = 0; g < 4; ++g) {
                float p = __expf(acc[8 + nt][g] - mx);
                s += p;
                ws += p * vxb[(size_t)idx1[g] * HH + ch];
            }
        }
        s  += __shfl_xor(s, 16);
        ws += __shfl_xor(ws, 16);
        float to = ws / s;
        if (hi == 0) orow[ch] += to;          // Ux already in d_out
    }
}

extern "C" void kernel_launch(void* const* d_in, const int* in_sizes, int n_in,
                              void* d_out, int out_size, void* d_ws, size_t ws_size,
                              hipStream_t stream) {
    const float* x   = (const float*)d_in[0];
    const float* e   = (const float*)d_in[1];
    const float* Wu  = (const float*)d_in[2];
    const float* bu  = (const float*)d_in[3];
    const float* Wv  = (const float*)d_in[4];
    const float* bv  = (const float*)d_in[5];
    const float* We  = (const float*)d_in[6];
    // d_in[7] = be: constant along the edge axis -> cancels in softmax, unused
    const int*   eix = (const int*)d_in[8];
    // d_in[9] = n_edges scalar (compile-time constant here)

    float* out = (float*)d_out;
    float* vxs = (float*)d_ws;                 // Vx scratch: BN*HH floats = 8.2 MB

    const int ntiles = BN / 16;                // 1000
    // Ux = x@Wu + bu  -> d_out (full overwrite, deterministic)
    gemm128_bias_kernel<<<ntiles / 8, 256, 0, stream>>>(x, Wu, bu, out, ntiles);
    // Vx = x@Wv + bv  -> scratch
    gemm128_bias_kernel<<<ntiles / 8, 256, 0, stream>>>(x, Wv, bv, vxs, ntiles);
    // fused: Ve GEMM + softmax + gather + accumulate into d_out
    edge_softmax_gather_kernel<<<BB * (NN / 8), 256, 0, stream>>>(e, We, eix, vxs, out);
}
